// Seq2Seq_78855599555231
// MI455X (gfx1250) — matrix operand order repaired and checked by reading, not compiled, not run on hardware
//
#include <hip/hip_runtime.h>
#include <hip/hip_bf16.h>

// Problem constants
#define B_  64
#define S_  64
#define T_  32
#define E_  512
#define H_  1024
#define V_  32000
#define H4_ 4096

typedef __bf16 bf16_t;
typedef __attribute__((ext_vector_type(16))) __bf16 v16bf;
typedef __attribute__((ext_vector_type(8)))  float  v8f;

// ---------------------------------------------------------------------------
// WMMA fragment loaders per CDNA5 ISA layouts (wave32)
// A: 16x32 bf16, lane(0-15)=row M, half=lane>>4 selects K sub-banks.
//    VGPR i holds K pair: k0 = (i>>2)*16 + half*8 + (i&3)*2
//    (per-lane bytes are two contiguous 16B runs -> compiler emits 2x b128)
// B: 32x16 bf16 = W^T. lane col = lane&15, kset = lane>>4 -> K runs
//    [kset*8, +8) and [16+kset*8, +8) from one row of W (row-major NxK), matching A.
// C/D: VGPR r -> M = half*8 + r, N = lane&15.
// ---------------------------------------------------------------------------
__device__ __forceinline__ v16bf s2s_load_A(const bf16_t* A, long long lda,
                                            int m0, long long kk, int lane) {
  const int half = lane >> 4, row = lane & 15;
  const bf16_t* base = A + (long long)(m0 + row) * lda + kk;
  union { unsigned u[8]; v16bf v; } ua;
#pragma unroll
  for (int p = 0; p < 8; ++p) {
    const int k0 = ((p >> 2) << 4) + (half << 3) + ((p & 3) << 1);
    ua.u[p] = *(const unsigned*)(base + k0);
  }
  return ua.v;
}

__device__ __forceinline__ v16bf s2s_load_B(const bf16_t* W, long long ldw,
                                            int n0, long long kk, int lane) {
  const int kset = lane >> 4, col = lane & 15;
  const bf16_t* base = W + (long long)(n0 + col) * ldw + kk + (kset << 3);
  union { uint4 q[2]; v16bf v; } ub;
  ub.q[0] = *(const uint4*)(base);
  ub.q[1] = *(const uint4*)(base + 16);
  return ub.v;
}

// ---------------------------------------------------------------------------
// Generic bf16 WMMA GEMM: C[M,N] = A1[M,K1] @ W1^T + A2[M,K2] @ W2^T + bias.
// One wave owns an (MT*16) x 16 output strip: one B fragment is reused by MT
// back-to-back WMMAs (MT=4 cuts B/L2 traffic 4x on the fc_W logits GEMM and
// keeps the XDL pipe fed). M % (16*MT) == 0, N % 16 == 0, K % 32 == 0.
// ---------------------------------------------------------------------------
template <int MT>
__global__ void s2s_wmma_gemm(const bf16_t* __restrict__ A1, long long lda1, int K1,
                              const bf16_t* __restrict__ A2, long long lda2, int K2,
                              const bf16_t* __restrict__ W1,
                              const bf16_t* __restrict__ W2,
                              const float* __restrict__ bias,
                              float* __restrict__ C, long long ldc,
                              int M, int N) {
  const int wave   = blockIdx.x * (blockDim.x >> 5) + (threadIdx.x >> 5);
  const int ntiles = N >> 4;
  const int mt = wave / ntiles;
  const int nt = wave - mt * ntiles;
  if (mt >= M / (16 * MT)) return;         // wave-uniform exit: EXEC stays all-1s
  const int lane = threadIdx.x & 31;
  const int m0 = mt * 16 * MT, n0 = nt << 4;

  v8f acc[MT] = {};
  for (long long kk = 0; kk < K1; kk += 32) {
    __builtin_prefetch((const void*)(W1 + (long long)(n0 + (lane & 15)) * K1 + kk + 64), 0, 0);
    v16bf b = s2s_load_B(W1, K1, n0, kk, lane);
#pragma unroll
    for (int i = 0; i < MT; ++i) {
      v16bf a = s2s_load_A(A1, lda1, m0 + 16 * i, kk, lane);
      acc[i] = __builtin_amdgcn_wmma_f32_16x16x32_bf16(false, a, false, b,
                                                       (short)0, acc[i], false, false);
    }
  }
  if (A2) {
    for (long long kk = 0; kk < K2; kk += 32) {
      v16bf b = s2s_load_B(W2, K2, n0, kk, lane);
#pragma unroll
      for (int i = 0; i < MT; ++i) {
        v16bf a = s2s_load_A(A2, lda2, m0 + 16 * i, kk, lane);
        acc[i] = __builtin_amdgcn_wmma_f32_16x16x32_bf16(false, a, false, b,
                                                         (short)0, acc[i], false, false);
      }
    }
  }

  const int half = lane >> 4, col = lane & 15;
  const float bv = bias ? bias[n0 + col] : 0.0f;
#pragma unroll
  for (int i = 0; i < MT; ++i) {
#pragma unroll
    for (int r = 0; r < 8; ++r) {
      const int m = m0 + 16 * i + (half << 3) + r;
      C[(long long)m * ldc + n0 + col] = acc[i][r] + bv;
    }
  }
}

// ---------------------------------------------------------------------------
// LSTM elementwise activation: gates[B,4H] (i,f,g,o) + c -> h,c updates.
// ---------------------------------------------------------------------------
__device__ __forceinline__ float s2s_sig(float x) { return 1.0f / (1.0f + __expf(-x)); }

__global__ void s2s_lstm_act(const float* __restrict__ gates, int Hd,
                             float* __restrict__ c,
                             bf16_t* __restrict__ h_bf,
                             float* __restrict__ out_f32, long long out_f32_rs,
                             bf16_t* __restrict__ out_bf2, long long out_bf2_rs,
                             int nb) {
  const int idx = blockIdx.x * blockDim.x + threadIdx.x;
  if (idx >= nb * Hd) return;
  const int b = idx / Hd, j = idx - b * Hd;
  const float* g = gates + (long long)b * 4 * Hd;
  const float ig = g[j], fg = g[Hd + j], gg = g[2 * Hd + j], og = g[3 * Hd + j];
  const float cn = s2s_sig(fg) * c[idx] + s2s_sig(ig) * tanhf(gg);
  const float hn = s2s_sig(og) * tanhf(cn);
  c[idx]    = cn;
  h_bf[idx] = (bf16_t)hn;
  if (out_f32) out_f32[(long long)b * out_f32_rs + j] = hn;
  if (out_bf2) out_bf2[(long long)b * out_bf2_rs + j] = (bf16_t)hn;
}

// ---------------------------------------------------------------------------
// Attention score: one wave per (b,s): score = sum_h v[h]*tanh(q[b,h]+P[b,s,h])
// ---------------------------------------------------------------------------
__global__ void s2s_attn_scores(const float* __restrict__ q,
                                const float* __restrict__ P,
                                const float* __restrict__ v,
                                float* __restrict__ scores) {
  const int wave = blockIdx.x * (blockDim.x >> 5) + (threadIdx.x >> 5);
  const int b = wave / S_, s = wave - b * S_;
  const int lane = threadIdx.x & 31;
  const float* qb = q + (long long)b * H_;
  const float* pb = P + ((long long)b * S_ + s) * H_;
  float sum = 0.0f;
  for (int h = lane; h < H_; h += 32) sum += v[h] * tanhf(qb[h] + pb[h]);
#pragma unroll
  for (int off = 16; off > 0; off >>= 1) sum += __shfl_down(sum, off, 32);
  if (lane == 0) scores[(long long)b * S_ + s] = sum;
}

// ---------------------------------------------------------------------------
// Softmax over S + context = a @ enc_out.  One block per batch row.
// ---------------------------------------------------------------------------
__global__ void s2s_softmax_ctx(const float* __restrict__ scores,
                                const float* __restrict__ enc_out,
                                bf16_t* __restrict__ rnn_in,
                                bf16_t* __restrict__ hcat) {
  const int b = blockIdx.x;
  __shared__ float a[S_];
  const int tid = threadIdx.x;
  if (tid < S_) a[tid] = scores[(long long)b * S_ + tid];
  __syncthreads();
  if (tid == 0) {
    float mx = a[0];
    for (int i = 1; i < S_; ++i) mx = fmaxf(mx, a[i]);
    float sm = 0.0f;
    for (int i = 0; i < S_; ++i) { a[i] = __expf(a[i] - mx); sm += a[i]; }
    const float inv = 1.0f / sm;
    for (int i = 0; i < S_; ++i) a[i] *= inv;
  }
  __syncthreads();
  for (int h = tid; h < H_; h += blockDim.x) {
    const float* eb = enc_out + (long long)b * S_ * H_ + h;
    float acc = 0.0f;
    for (int s = 0; s < S_; ++s) acc += a[s] * eb[(long long)s * H_];
    rnn_in[(long long)b * (E_ + H_) + E_ + h] = (bf16_t)acc;
    hcat[(long long)b * (2 * H_) + H_ + h]    = (bf16_t)acc;
  }
}

// ---------------------------------------------------------------------------
// Utility kernels: conversions, gathers, zero-fills
// ---------------------------------------------------------------------------
__global__ void s2s_cvt_bf16(const float* __restrict__ src, long long src_rs,
                             bf16_t* __restrict__ dst, long long dst_rs,
                             int rowlen, long long total) {
  const long long idx = (long long)blockIdx.x * blockDim.x + threadIdx.x;
  if (idx >= total) return;
  const long long r = idx / rowlen;
  const int cc = (int)(idx - r * rowlen);
  dst[r * dst_rs + cc] = (bf16_t)src[r * src_rs + cc];
}

__global__ void s2s_add2(const float* a, const float* b, float* o, int n) {
  const int i = blockIdx.x * blockDim.x + threadIdx.x;
  if (i < n) o[i] = a[i] + b[i];
}

__global__ void s2s_zero_f32(float* p, long long rstride, int rowlen, long long total) {
  const long long idx = (long long)blockIdx.x * blockDim.x + threadIdx.x;
  if (idx >= total) return;
  const long long r = idx / rowlen;
  p[r * rstride + (idx - r * rowlen)] = 0.0f;
}

__global__ void s2s_zero_bf16(bf16_t* p, long long n) {
  const long long i = (long long)blockIdx.x * blockDim.x + threadIdx.x;
  if (i < n) p[i] = (bf16_t)0.0f;
}

__global__ void s2s_gather_src(const int* __restrict__ src,
                               const float* __restrict__ emb,
                               bf16_t* __restrict__ x_b, long long total) {
  const long long idx = (long long)blockIdx.x * blockDim.x + threadIdx.x;
  if (idx >= total) return;
  const long long row = idx / E_;
  const int e = (int)(idx - row * E_);
  const int tok = src[row];
  x_b[idx] = (bf16_t)emb[(long long)tok * E_ + e];
}

__global__ void s2s_gather_dec(const int* __restrict__ tgt, int t,
                               const float* __restrict__ emb,
                               bf16_t* __restrict__ rnn_in) {
  const int idx = blockIdx.x * blockDim.x + threadIdx.x;
  if (idx >= B_ * E_) return;
  const int b = idx / E_, e = idx - b * E_;
  const int tok = tgt[(long long)b * T_ + t];
  rnn_in[(long long)b * (E_ + H_) + e] = (bf16_t)emb[(long long)tok * E_ + e];
}

// ---------------------------------------------------------------------------
// Host side
// ---------------------------------------------------------------------------
template <int MT>
static inline void launch_gemm(hipStream_t st,
                               const bf16_t* A1, long long lda1, int K1,
                               const bf16_t* A2, long long lda2, int K2,
                               const bf16_t* W1, const bf16_t* W2,
                               const float* bias, float* C, long long ldc,
                               int M, int N) {
  const int waves = (M / (16 * MT)) * (N / 16);
  const int blocks = (waves + 7) / 8;
  s2s_wmma_gemm<MT><<<blocks, 256, 0, st>>>(A1, lda1, K1, A2, lda2, K2,
                                            W1, W2, bias, C, ldc, M, N);
}

extern "C" void kernel_launch(void* const* d_in, const int* in_sizes, int n_in,
                              void* d_out, int out_size, void* d_ws, size_t ws_size,
                              hipStream_t stream) {
  (void)in_sizes; (void)n_in; (void)out_size; (void)ws_size;

  const int*   src     = (const int*)d_in[0];
  const int*   tgt     = (const int*)d_in[1];
  const float* enc_emb = (const float*)d_in[2];
  const float* dec_emb = (const float*)d_in[3];
  const float* enc_Wih = (const float*)d_in[4];
  const float* enc_Whh = (const float*)d_in[5];
  const float* enc_bih = (const float*)d_in[6];
  const float* enc_bhh = (const float*)d_in[7];
  const float* dec_Wih = (const float*)d_in[8];
  const float* dec_Whh = (const float*)d_in[9];
  const float* dec_bih = (const float*)d_in[10];
  const float* dec_bhh = (const float*)d_in[11];
  const float* attn_W  = (const float*)d_in[12];
  const float* attn_b  = (const float*)d_in[13];
  const float* vvec    = (const float*)d_in[14];
  const float* fc_W    = (const float*)d_in[15];
  const float* fc_b    = (const float*)d_in[16];
  float* out = (float*)d_out;

  // ---- workspace carve-out (256B aligned slices) ----
  char* ws = (char*)d_ws;
  size_t off = 0;
  auto take = [&](size_t bytes) -> char* {
    char* p = ws + off;
    off = (off + bytes + 255) & ~(size_t)255;
    return p;
  };
  bf16_t* fcW_b   = (bf16_t*)take((size_t)V_ * 2 * H_ * 2);
  bf16_t* eWih_b  = (bf16_t*)take((size_t)H4_ * E_ * 2);
  bf16_t* eWhh_b  = (bf16_t*)take((size_t)H4_ * H_ * 2);
  bf16_t* dWih_b  = (bf16_t*)take((size_t)H4_ * (E_ + H_) * 2);
  bf16_t* dWhh_b  = (bf16_t*)take((size_t)H4_ * H_ * 2);
  bf16_t* Wh_b    = (bf16_t*)take((size_t)H_ * H_ * 2);
  bf16_t* We_b    = (bf16_t*)take((size_t)H_ * H_ * 2);
  bf16_t* x_b     = (bf16_t*)take((size_t)B_ * S_ * E_ * 2);
  float*  encO_f  = (float*) take((size_t)B_ * S_ * H_ * 4);
  bf16_t* encO_b  = (bf16_t*)take((size_t)B_ * S_ * H_ * 2);
  float*  P       = (float*) take((size_t)B_ * S_ * H_ * 4);
  float*  gates   = (float*) take((size_t)B_ * H4_ * 4);
  bf16_t* h_b     = (bf16_t*)take((size_t)B_ * H_ * 2);
  float*  cbuf    = (float*) take((size_t)B_ * H_ * 4);
  float*  q       = (float*) take((size_t)B_ * H_ * 4);
  float*  scores  = (float*) take((size_t)B_ * S_ * 4);
  bf16_t* rnn_b   = (bf16_t*)take((size_t)B_ * (E_ + H_) * 2);
  bf16_t* hcat_b  = (bf16_t*)take((size_t)B_ * 2 * H_ * 2);
  float*  biasE   = (float*) take((size_t)H4_ * 4);
  float*  biasD   = (float*) take((size_t)H4_ * 4);

  const int TB = 256;
  auto gridFor = [](long long n) { return (unsigned)((n + 255) / 256); };

  // ---- one-time weight conversion to bf16 (makes fc_W L2-resident) ----
  {
    long long n;
    n = (long long)H4_ * E_;
    s2s_cvt_bf16<<<gridFor(n), TB, 0, stream>>>(enc_Wih, n, eWih_b, n, E_, n);
    n = (long long)H4_ * H_;
    s2s_cvt_bf16<<<gridFor(n), TB, 0, stream>>>(enc_Whh, n, eWhh_b, n, H_, n);
    n = (long long)H4_ * (E_ + H_);
    s2s_cvt_bf16<<<gridFor(n), TB, 0, stream>>>(dec_Wih, n, dWih_b, n, (E_ + H_), n);
    n = (long long)H4_ * H_;
    s2s_cvt_bf16<<<gridFor(n), TB, 0, stream>>>(dec_Whh, n, dWhh_b, n, H_, n);
    // attn_W is [H, 2H]: Wh = [:, :H], We = [:, H:]
    n = (long long)H_ * H_;
    s2s_cvt_bf16<<<gridFor(n), TB, 0, stream>>>(attn_W,      2 * H_, Wh_b, H_, H_, n);
    s2s_cvt_bf16<<<gridFor(n), TB, 0, stream>>>(attn_W + H_, 2 * H_, We_b, H_, H_, n);
    n = (long long)V_ * 2 * H_;
    s2s_cvt_bf16<<<gridFor(n), TB, 0, stream>>>(fc_W, n, fcW_b, n, 2 * H_, n);
  }
  s2s_add2<<<gridFor(H4_), TB, 0, stream>>>(enc_bih, enc_bhh, biasE, H4_);
  s2s_add2<<<gridFor(H4_), TB, 0, stream>>>(dec_bih, dec_bhh, biasD, H4_);

  // ---- init state + t=0 output slice ----
  s2s_zero_f32 <<<gridFor((long long)B_ * H_), TB, 0, stream>>>(cbuf, (long long)B_ * H_, B_ * H_, (long long)B_ * H_);
  s2s_zero_bf16<<<gridFor((long long)B_ * H_), TB, 0, stream>>>(h_b, (long long)B_ * H_);
  s2s_zero_f32 <<<gridFor((long long)B_ * V_), TB, 0, stream>>>(out, (long long)T_ * V_, V_, (long long)B_ * V_);

  // ---- encoder ----
  s2s_gather_src<<<gridFor((long long)B_ * S_ * E_), TB, 0, stream>>>(src, enc_emb, x_b, (long long)B_ * S_ * E_);

  for (int t = 0; t < S_; ++t) {
    launch_gemm<2>(stream,
                   x_b + (long long)t * E_, (long long)S_ * E_, E_,
                   h_b, H_, H_,
                   eWih_b, eWhh_b, biasE,
                   gates, H4_, B_, H4_);
    s2s_lstm_act<<<gridFor((long long)B_ * H_), TB, 0, stream>>>(
        gates, H_, cbuf, h_b,
        encO_f + (long long)t * H_, (long long)S_ * H_,
        encO_b + (long long)t * H_, (long long)S_ * H_, B_);
  }

  // enc_proj (+ attn_b fused): P[b,s,k] = enc_out @ We^T + attn_b
  launch_gemm<4>(stream, encO_b, H_, H_, nullptr, 0, 0,
                 We_b, nullptr, attn_b, P, H_, B_ * S_, H_);

  // ---- decoder ----
  for (int t = 0; t < T_ - 1; ++t) {
    s2s_gather_dec<<<gridFor((long long)B_ * E_), TB, 0, stream>>>(tgt, t, dec_emb, rnn_b);

    // q = h @ Wh^T
    launch_gemm<1>(stream, h_b, H_, H_, nullptr, 0, 0,
                   Wh_b, nullptr, nullptr, q, H_, B_, H_);

    s2s_attn_scores<<<(B_ * S_) / 8, 256, 0, stream>>>(q, P, vvec, scores);
    s2s_softmax_ctx<<<B_, 256, 0, stream>>>(scores, encO_f, rnn_b, hcat_b);

    // gates = [emb|ctx] @ dec_Wih^T + h @ dec_Whh^T + bias
    launch_gemm<2>(stream,
                   rnn_b, (long long)(E_ + H_), E_ + H_,
                   h_b, H_, H_,
                   dWih_b, dWhh_b, biasD,
                   gates, H4_, B_, H4_);

    s2s_lstm_act<<<gridFor((long long)B_ * H_), TB, 0, stream>>>(
        gates, H_, cbuf, h_b,
        nullptr, 0,
        hcat_b, (long long)(2 * H_), B_);

    // logits -> d_out[:, t+1, :]  (fc_W bf16 is L2-resident; 4 WMMAs per B frag)
    launch_gemm<4>(stream, hcat_b, (long long)(2 * H_), 2 * H_, nullptr, 0, 0,
                   fcW_b, nullptr, fc_b,
                   out + (long long)(t + 1) * V_, (long long)T_ * V_, B_, V_);
  }
}
